// GCNNet_12695923327677
// MI455X (gfx1250) — matrix operand order repaired and checked by reading, not compiled, not run on hardware
//
#include <hip/hip_runtime.h>
#include <math.h>

// GCN layer for MI455X (gfx1250), wave32 + WMMA.
// Pipeline: init(agg=b1, deg=0) -> degree atomics -> sqrt -> GEMM1 (WMMA f16-split)
//           -> edge scatter (float atomics, L2 resident) -> GEMM2 (WMMA f16-split).

typedef __attribute__((ext_vector_type(16))) _Float16 v16h;
typedef __attribute__((ext_vector_type(8)))  _Float16 v8h;
typedef __attribute__((ext_vector_type(4)))  _Float16 v4h;
typedef __attribute__((ext_vector_type(8)))  float    v8f;
typedef __attribute__((ext_vector_type(4)))  float    v4f;

#define IN_C  512
#define HID_C 128
#define OUT_C 16

// ---------------------------------------------------------------- utilities
__device__ __forceinline__ void split16(const float* xv, v16h& hi, v16h& lo) {
#pragma unroll
  for (int j = 0; j < 16; ++j) {
    float f = xv[j];
    _Float16 hj = (_Float16)f;
    hi[j] = hj;
    lo[j] = (_Float16)(f - (float)hj);
  }
}

__device__ __forceinline__ v16h lds_frag(const _Float16* p) {
  // two 16B-aligned LDS b128 loads combined into a 16-half fragment
  v8h a = *(const v8h*)p;
  v8h b = *(const v8h*)(p + 8);
  return __builtin_shufflevector(a, b, 0, 1, 2, 3, 4, 5, 6, 7,
                                 8, 9, 10, 11, 12, 13, 14, 15);
}

// ---------------------------------------------------------------- setup kernels
__global__ __launch_bounds__(256) void init_kernel(float* __restrict__ agg,
                                                   float* __restrict__ deg,
                                                   const float* __restrict__ b1,
                                                   int n128, int n) {
  int i = blockIdx.x * 256 + threadIdx.x;
  if (i < n128) agg[i] = b1[i & (HID_C - 1)];  // fold conv bias into agg init
  if (i < n)    deg[i] = 0.0f;
}

__global__ __launch_bounds__(256) void degree_kernel(const int* __restrict__ col,
                                                     float* __restrict__ deg, int e) {
  int i = blockIdx.x * 256 + threadIdx.x;
  if (i < e) atomicAdd(&deg[col[i]], 1.0f);
}

__global__ __launch_bounds__(256) void sqrt_kernel(const float* __restrict__ deg,
                                                   float* __restrict__ d, int n) {
  int i = blockIdx.x * 256 + threadIdx.x;
  if (i < n) d[i] = sqrtf(deg[i]);  // faithful: deg.pow(0.5), NOT inv-sqrt
}

// ---------------------------------------------------------------- GEMM1: h = x @ W1^T
// Block = 256 threads = 8 wave32; each wave -> 16 rows x 128 cols of h.
// W1 chunked over K (64 wide) into LDS as f16 hi/lo, padded stride 72 halves.
#define G1_LDSW 72

__global__ __launch_bounds__(256) void gemm1_kernel(const float* __restrict__ x,
                                                    const float* __restrict__ W1,
                                                    float* __restrict__ h, int n) {
  __shared__ _Float16 sWhi[HID_C * G1_LDSW];
  __shared__ _Float16 sWlo[HID_C * G1_LDSW];
  const int tid  = threadIdx.x;
  const int lane = tid & 31;
  const int wave = tid >> 5;
  const int row0 = (blockIdx.x * 8 + wave) * 16;
  const int m    = lane & 15;         // A: matrix row / B: matrix col / D: col
  const int og   = (lane >> 4) * 16;  // A fragment K-group offset
  const int bk   = (lane >> 4) * 16;  // B fragment K-group offset
  const bool active = row0 < n;

  const v8f vzero = {0.f, 0.f, 0.f, 0.f, 0.f, 0.f, 0.f, 0.f};
  v8f acc[8];
#pragma unroll
  for (int t = 0; t < 8; ++t) acc[t] = vzero;

  for (int kc = 0; kc < IN_C; kc += 64) {
    __syncthreads();
    // stage W1[:, kc:kc+64] into LDS as f16 hi/lo (128 rows x 64 cols)
    for (int i = tid; i < HID_C * 16; i += 256) {
      int r  = i >> 4;
      int c4 = (i & 15) << 2;
      v4f v = *(const v4f*)(W1 + (size_t)r * IN_C + kc + c4);
      v4h hi4, lo4;
#pragma unroll
      for (int j = 0; j < 4; ++j) {
        _Float16 hj = (_Float16)v[j];
        hi4[j] = hj;
        lo4[j] = (_Float16)(v[j] - (float)hj);
      }
      *(v4h*)(sWhi + r * G1_LDSW + c4) = hi4;
      *(v4h*)(sWlo + r * G1_LDSW + c4) = lo4;
    }
    __syncthreads();

    if (active) {
#pragma unroll
      for (int ki = 0; ki < 64; ki += 32) {
        // A fragment: 16 floats of x for this lane, split to f16 hi/lo
        const float* xp = x + (size_t)(row0 + m) * IN_C + kc + ki + og;
        float xv[16];
        *(v4f*)(xv + 0)  = *(const v4f*)(xp + 0);
        *(v4f*)(xv + 4)  = *(const v4f*)(xp + 4);
        *(v4f*)(xv + 8)  = *(const v4f*)(xp + 8);
        *(v4f*)(xv + 12) = *(const v4f*)(xp + 12);
        v16h ahi, alo;
        split16(xv, ahi, alo);
#pragma unroll
        for (int t = 0; t < 8; ++t) {
          const _Float16* bp = sWhi + (t * 16 + m) * G1_LDSW + ki + bk;
          v16h bhi = lds_frag(bp);
          v16h blo = lds_frag(sWlo + (t * 16 + m) * G1_LDSW + ki + bk);
          // 3-term f16 split: hi*hi + lo*hi + hi*lo  (~fp32 accuracy)
          acc[t] = __builtin_amdgcn_wmma_f32_16x16x32_f16(
              false, ahi, false, bhi, (short)0, acc[t], false, false);
          acc[t] = __builtin_amdgcn_wmma_f32_16x16x32_f16(
              false, alo, false, bhi, (short)0, acc[t], false, false);
          acc[t] = __builtin_amdgcn_wmma_f32_16x16x32_f16(
              false, ahi, false, blo, (short)0, acc[t], false, false);
        }
      }
    }
  }

  if (active) {
    const int mo = (lane >> 4) * 8;  // C/D layout: VGPR v -> M = v (+8 upper half)
#pragma unroll
    for (int t = 0; t < 8; ++t) {
#pragma unroll
      for (int v = 0; v < 8; ++v)
        h[(size_t)(row0 + mo + v) * HID_C + t * 16 + m] = acc[t][v];
    }
  }
}

// ---------------------------------------------------------------- edge scatter
// One wave per edge; lane j handles channels 4j..4j+3 (coalesced 512B gather),
// 4 global f32 atomics into L2-resident agg.
__global__ __launch_bounds__(256) void scatter_kernel(const int* __restrict__ rowi,
                                                      const int* __restrict__ coli,
                                                      const float* __restrict__ dvec,
                                                      const float* __restrict__ h,
                                                      float* __restrict__ agg, int e) {
  int ei = blockIdx.x * 8 + (threadIdx.x >> 5);
  if (ei >= e) return;
  int lane = threadIdx.x & 31;
  int r = rowi[ei];
  int c = coli[ei];
  float norm = dvec[r] * dvec[c];
  v4f v = *(const v4f*)(h + (size_t)r * HID_C + lane * 4);
  float* ap = agg + (size_t)c * HID_C + lane * 4;
  atomicAdd(ap + 0, norm * v[0]);
  atomicAdd(ap + 1, norm * v[1]);
  atomicAdd(ap + 2, norm * v[2]);
  atomicAdd(ap + 3, norm * v[3]);
}

// ---------------------------------------------------------------- GEMM2: out = agg @ W2^T + b2
#define G2_LDSW 136

__global__ __launch_bounds__(256) void gemm2_kernel(const float* __restrict__ agg,
                                                    const float* __restrict__ W2,
                                                    const float* __restrict__ b2,
                                                    float* __restrict__ out, int n) {
  __shared__ _Float16 sWhi[OUT_C * G2_LDSW];
  __shared__ _Float16 sWlo[OUT_C * G2_LDSW];
  const int tid = threadIdx.x;
  // stage all of W2 (16x128) as f16 hi/lo
  for (int i = tid; i < OUT_C * 32; i += 256) {
    int r  = i >> 5;
    int c4 = (i & 31) << 2;
    v4f v = *(const v4f*)(W2 + (size_t)r * HID_C + c4);
    v4h hi4, lo4;
#pragma unroll
    for (int j = 0; j < 4; ++j) {
      _Float16 hj = (_Float16)v[j];
      hi4[j] = hj;
      lo4[j] = (_Float16)(v[j] - (float)hj);
    }
    *(v4h*)(sWhi + r * G2_LDSW + c4) = hi4;
    *(v4h*)(sWlo + r * G2_LDSW + c4) = lo4;
  }
  __syncthreads();

  const int lane = tid & 31;
  const int wave = tid >> 5;
  const int row0 = (blockIdx.x * 8 + wave) * 16;
  if (row0 >= n) return;
  const int m  = lane & 15;
  const int og = (lane >> 4) * 16;
  const int bk = (lane >> 4) * 16;

  v8f acc = {0.f, 0.f, 0.f, 0.f, 0.f, 0.f, 0.f, 0.f};
#pragma unroll
  for (int ki = 0; ki < HID_C; ki += 32) {
    const float* ap = agg + (size_t)(row0 + m) * HID_C + ki + og;
    float xv[16];
    *(v4f*)(xv + 0)  = *(const v4f*)(ap + 0);
    *(v4f*)(xv + 4)  = *(const v4f*)(ap + 4);
    *(v4f*)(xv + 8)  = *(const v4f*)(ap + 8);
    *(v4f*)(xv + 12) = *(const v4f*)(ap + 12);
    v16h ahi, alo;
    split16(xv, ahi, alo);
    v16h bhi = lds_frag(sWhi + m * G2_LDSW + ki + bk);
    v16h blo = lds_frag(sWlo + m * G2_LDSW + ki + bk);
    acc = __builtin_amdgcn_wmma_f32_16x16x32_f16(false, ahi, false, bhi,
                                                 (short)0, acc, false, false);
    acc = __builtin_amdgcn_wmma_f32_16x16x32_f16(false, alo, false, bhi,
                                                 (short)0, acc, false, false);
    acc = __builtin_amdgcn_wmma_f32_16x16x32_f16(false, ahi, false, blo,
                                                 (short)0, acc, false, false);
  }

  float bias = b2[m];
  const int mo = (lane >> 4) * 8;
#pragma unroll
  for (int v = 0; v < 8; ++v)
    out[(size_t)(row0 + mo + v) * OUT_C + m] = acc[v] + bias;
}

// ---------------------------------------------------------------- launcher
extern "C" void kernel_launch(void* const* d_in, const int* in_sizes, int n_in,
                              void* d_out, int out_size, void* d_ws, size_t ws_size,
                              hipStream_t stream) {
  const float* x  = (const float*)d_in[0];
  const int*   ei = (const int*)d_in[1];   // [2, E] int32
  const float* W1 = (const float*)d_in[2]; // [128, 512]
  const float* b1 = (const float*)d_in[3]; // [128]
  const float* W2 = (const float*)d_in[4]; // [16, 128]
  const float* b2 = (const float*)d_in[5]; // [16]
  float* out = (float*)d_out;

  const int N = in_sizes[0] / IN_C;
  const int E = in_sizes[1] / 2;
  const int* rowi = ei;
  const int* coli = ei + E;

  // workspace layout: h [N*128] | agg [N*128] | deg [N] | d [N]
  float* h    = (float*)d_ws;
  float* agg  = h + (size_t)N * HID_C;
  float* deg  = agg + (size_t)N * HID_C;
  float* dvec = deg + N;

  const int n128 = N * HID_C;
  init_kernel<<<(n128 + 255) / 256, 256, 0, stream>>>(agg, deg, b1, n128, N);
  degree_kernel<<<(E + 255) / 256, 256, 0, stream>>>(coli, deg, E);
  sqrt_kernel<<<(N + 255) / 256, 256, 0, stream>>>(deg, dvec, N);
  gemm1_kernel<<<(N + 127) / 128, 256, 0, stream>>>(x, W1, h, N);
  scatter_kernel<<<(E + 7) / 8, 256, 0, stream>>>(rowi, coli, dvec, h, agg, E);
  gemm2_kernel<<<(N + 127) / 128, 256, 0, stream>>>(agg, W2, b2, out, N);
}
